// MoEUQ_network_36498632081500
// MI455X (gfx1250) — compile-verified, matrix operand order checked by reading, not run on hardware
//
#include <hip/hip_runtime.h>
#include <hip/hip_bf16.h>
#include <cstdint>

// ---------------------------------------------------------------------------
// CDNA5 (gfx1250) implementation: f16 WMMA (fp32 accumulate) for all GEMMs.
// wave32, 8 waves / 256-thread block.
//
// LSTM recurrence is split: the non-recurrent x@Wih.T is a fully parallel
// WMMA pre-GEMM writing C-fragments; the sequential kernel only does the
// h@Whh.T WMMAs per step (half the critical-path matrix work), with
// hardware v_tanh_f32 for the gate nonlinearities.
// ---------------------------------------------------------------------------

typedef __attribute__((ext_vector_type(16))) _Float16 v16h;
typedef __attribute__((ext_vector_type(8)))  _Float16 v8h;
typedef __attribute__((ext_vector_type(8)))  float    v8f;

#define BDIM 256
#define LDIM 128
#define DDIM 128
#define NEXP 8
#define TTOK (BDIM * LDIM)   // 32768 tokens

// Load one 16-half A/B fragment slice for v_wmma_f32_16x16x32_f16.
// Lanes 0-15 hold K {k0..k0+7} in halves 0..7 and {k0+16..k0+23} in halves
// 8..15; lanes 16-31 get the +8 shifted octets (folded into `half16`).
__device__ __forceinline__ v16h load_frag16(const _Float16* p) {
  v8h lo = *reinterpret_cast<const v8h*>(p);
  v8h hi = *reinterpret_cast<const v8h*>(p + 16);
  return __builtin_shufflevector(lo, hi, 0, 1, 2, 3, 4, 5, 6, 7,
                                         8, 9, 10, 11, 12, 13, 14, 15);
}

__device__ __forceinline__ v8f wmma16(v16h a, v16h b, v8f c) {
  return __builtin_amdgcn_wmma_f32_16x16x32_f16(
      /*neg_a=*/false, a, /*neg_b=*/false, b,
      /*c_mod=*/(short)0, c, /*reuse_a=*/false, /*reuse_b=*/false);
}

// gfx1250 hardware transcendental tanh (v_tanh_f32) with safe fallback.
__device__ __forceinline__ float fast_tanh(float x) {
#if __has_builtin(__builtin_amdgcn_tanhf)
  return __builtin_amdgcn_tanhf(x);
#elif __has_builtin(__builtin_amdgcn_tanh_f32)
  return __builtin_amdgcn_tanh_f32(x);
#else
  return tanhf(x);
#endif
}
// sigmoid(x) = 0.5*tanh(x/2) + 0.5 : single TRANS op + FMA
__device__ __forceinline__ float fast_sigmoid(float x) {
  return fmaf(0.5f, fast_tanh(0.5f * x), 0.5f);
}
__device__ __forceinline__ float softplusf_(float x) {
  if (x > 20.0f)  return x;
  if (x < -20.0f) return __expf(x);
  return log1pf(__expf(x));
}

// ---------------------------------------------------------------------------
// Small utility kernels
// ---------------------------------------------------------------------------
__global__ __launch_bounds__(256) void f2h_kernel(const float* __restrict__ s,
                                                  _Float16* __restrict__ d, int n) {
  int i = blockIdx.x * 256 + threadIdx.x;
  if (i < n) d[i] = (_Float16)s[i];
}

__global__ __launch_bounds__(256) void addbias_kernel(const float* __restrict__ a,
                                                      const float* __restrict__ b,
                                                      float* __restrict__ o, int n) {
  int i = blockIdx.x * 256 + threadIdx.x;
  if (i < n) o[i] = a[i] + b[i];
}

// ---------------------------------------------------------------------------
// rec_in = relu(concat(seg_emb[xs], slice_emb[ts]) @ W_all.T + b_all) -> f16
// ---------------------------------------------------------------------------
__global__ __launch_bounds__(128) void recin_kernel(
    const int* __restrict__ xs, const int* __restrict__ ts,
    const float* __restrict__ seg_emb, const float* __restrict__ slice_emb,
    const float* __restrict__ W_all, const float* __restrict__ b_all,
    _Float16* __restrict__ rec) {
  const int t = blockIdx.x;
  const int d = threadIdx.x;
  const int b = t / LDIM;
  const int s  = xs[t];
  const int sl = ts[b];
  const float* se = seg_emb + (size_t)s * 20;
  const float* le = slice_emb + (size_t)sl * 20;
  const float* w  = W_all + (size_t)d * 40;
  float acc = b_all[d];
#pragma unroll
  for (int k = 0; k < 20; ++k) acc += se[k] * w[k];
#pragma unroll
  for (int k = 0; k < 20; ++k) acc += le[k] * w[20 + k];
  acc = fmaxf(acc, 0.0f);
  rec[(size_t)t * DDIM + d] = (_Float16)acc;
}

// ---------------------------------------------------------------------------
// deep = relu(bn(concat(ts, o_emb, d_emb) @ W_dm.T + b_dm))
// ---------------------------------------------------------------------------
__global__ __launch_bounds__(128) void deep_kernel(
    const int* __restrict__ od, const int* __restrict__ ts,
    const float* __restrict__ node_emb, const float* __restrict__ W_dm,
    const float* __restrict__ b_dm, const float* __restrict__ g_dm,
    const float* __restrict__ beta_dm, float* __restrict__ deep) {
  const int b = blockIdx.x;
  const int d = threadIdx.x;
  const int o  = od[b * 2 + 0];
  const int dd = od[b * 2 + 1];
  const float* oe = node_emb + (size_t)o * 20;
  const float* de = node_emb + (size_t)dd * 20;
  const float* w  = W_dm + (size_t)d * 41;
  float acc = b_dm[d] + w[0] * (float)ts[b];
#pragma unroll
  for (int k = 0; k < 20; ++k) acc += oe[k] * w[1 + k];
#pragma unroll
  for (int k = 0; k < 20; ++k) acc += de[k] * w[21 + k];
  const float inv = rsqrtf(1.0f + 1e-5f);
  acc = acc * (g_dm[d] * inv) + beta_dm[d];
  deep[(size_t)b * DDIM + d] = fmaxf(acc, 0.0f);
}

// ---------------------------------------------------------------------------
// Pre-GEMM: xw = x @ Wih.T for ALL timesteps (fully parallel WMMA).
// Output stored directly in WMMA C-fragment layout:
//   xw[tile_b][t][n(512)][m(16)]  (v8f per lane, 32B aligned)
// Grid: (L, B/16), 256 threads (8 waves); wave w -> cols [64w, 64w+64).
// ---------------------------------------------------------------------------
__global__ __launch_bounds__(256) void xw_kernel(
    const _Float16* __restrict__ xin,  // [B*L, 128] f16
    const _Float16* __restrict__ Wih,  // [512, 128] f16
    float* __restrict__ xw)            // [B/16][L][512][16] f32
{
  const int wave = threadIdx.x >> 5;
  const int lane = threadIdx.x & 31;
  const int l15  = lane & 15;
  const int half16 = (lane >> 4) << 3;
  const int t  = blockIdx.x;
  const int tb = blockIdx.y;
  const int b0 = tb * 16;

  const _Float16* xrow = xin + ((size_t)(b0 + l15) * LDIM + t) * DDIM + half16;
  v16h a0 = load_frag16(xrow + 0);
  v16h a1 = load_frag16(xrow + 32);
  v16h a2 = load_frag16(xrow + 64);
  v16h a3 = load_frag16(xrow + 96);

  float* obase = xw + ((size_t)tb * LDIM + t) * 512 * 16;
#pragma unroll
  for (int s = 0; s < 4; ++s) {
    const int n0 = wave * 64 + s * 16;
    const _Float16* pb = Wih + (size_t)(n0 + l15) * DDIM + half16;
    v8f acc = {};
    acc = wmma16(a0, load_frag16(pb + 0),  acc);
    acc = wmma16(a1, load_frag16(pb + 32), acc);
    acc = wmma16(a2, load_frag16(pb + 64), acc);
    acc = wmma16(a3, load_frag16(pb + 96), acc);
    *reinterpret_cast<v8f*>(obase + (size_t)(n0 + l15) * 16 + half16) = acc;
  }
}

// ---------------------------------------------------------------------------
// Sequential LSTM recurrence. Grid: B/16 blocks; block owns 16 batch rows
// for all L steps. Per step each wave: 4 gate tiles, accumulator seeded
// from the precomputed x@Wih.T fragments, + 16 h@Whh.T WMMAs. Cell state
// lives in the v8f accumulator tile; h lives in LDS (f16).
// ---------------------------------------------------------------------------
__global__ __launch_bounds__(256) void lstm_seq_kernel(
    const float*    __restrict__ xw,   // [B/16][L][512][16] f32 (C-fragments)
    const _Float16* __restrict__ Whh,  // [512, 128] f16
    const float*    __restrict__ bias, // [512] = bih + bhh
    _Float16*       __restrict__ hout) // [B*L, 128] f16
{
  __shared__ __align__(16) _Float16 h_lds[16 * DDIM];
  const int wave = threadIdx.x >> 5;
  const int lane = threadIdx.x & 31;
  const int l15  = lane & 15;
  const int half16 = (lane >> 4) << 3;
  const int b0 = blockIdx.x * 16;
  const int col = wave * 16 + l15;      // hidden column owned by this lane

  for (int i = threadIdx.x; i < 16 * DDIM; i += 256) h_lds[i] = (_Float16)0.0f;

  const float bi = bias[0 * DDIM + col];
  const float bf = bias[1 * DDIM + col];
  const float bg = bias[2 * DDIM + col];
  const float bo = bias[3 * DDIM + col];

  v8f cst = {};  // cell state tile (16 batch rows x 16 hidden cols)
  __syncthreads();

  const float* xwblk = xw + (size_t)blockIdx.x * LDIM * 512 * 16;

  for (int t = 0; t < LDIM; ++t) {
    const float* xwt = xwblk + (size_t)t * 512 * 16;
    const _Float16* hrow = &h_lds[l15 * DDIM + half16];
    v16h ah0 = load_frag16(hrow + 0);
    v16h ah1 = load_frag16(hrow + 32);
    v16h ah2 = load_frag16(hrow + 64);
    v16h ah3 = load_frag16(hrow + 96);
    if (t + 1 < LDIM)  // gfx1250 global_prefetch_b8 of next step's fragments
      __builtin_prefetch(xwblk + (size_t)(t + 1) * 512 * 16 +
                             (size_t)col * 16 + half16, 0, 3);
    __syncthreads();  // all waves captured h before it is overwritten

    auto gate = [&](int g) -> v8f {
      const int n0 = g * DDIM + wave * 16;
      v8f acc = *reinterpret_cast<const v8f*>(
          xwt + (size_t)(n0 + l15) * 16 + half16);  // x@Wih.T fragment
      const _Float16* pbh = Whh + (size_t)(n0 + l15) * DDIM + half16;
      acc = wmma16(ah0, load_frag16(pbh + 0),  acc);
      acc = wmma16(ah1, load_frag16(pbh + 32), acc);
      acc = wmma16(ah2, load_frag16(pbh + 64), acc);
      acc = wmma16(ah3, load_frag16(pbh + 96), acc);
      return acc;
    };
    v8f zi = gate(0);
    v8f zf = gate(1);
    v8f zg = gate(2);
    v8f zo = gate(3);

#pragma unroll
    for (int j = 0; j < 8; ++j) {
      float vi = fast_sigmoid(zi[j] + bi);
      float vf = fast_sigmoid(zf[j] + bf);
      float vg = fast_tanh(zg[j] + bg);
      float vo = fast_sigmoid(zo[j] + bo);
      float c  = vf * cst[j] + vi * vg;
      cst[j]   = c;
      float h  = vo * fast_tanh(c);
      const int m = j + half16;  // batch row within tile
      _Float16 hh = (_Float16)h;
      h_lds[m * DDIM + col] = hh;
      hout[((size_t)(b0 + m) * LDIM + t) * DDIM + col] = hh;
    }
    __syncthreads();  // h writes visible before next step's loads
  }
}

// ---------------------------------------------------------------------------
// Router: logits/noisy/top-2 gating (mask folded in) + per-block soft_gate
// partial sums (deterministic serial reduction, no atomics).
// ---------------------------------------------------------------------------
__global__ __launch_bounds__(256) void router_kernel(
    const _Float16* __restrict__ seq, const float* __restrict__ W_route,
    const float* __restrict__ b_route, const float* __restrict__ W_noise,
    const float* __restrict__ b_noise, const float* __restrict__ noise,
    const int* __restrict__ nseg, float* __restrict__ gating,
    float* __restrict__ loadpart) {
  __shared__ float smbuf[256 * NEXP];
  const int t = blockIdx.x * 256 + threadIdx.x;
  const int b = t / LDIM;
  const int l = t % LDIM;

  float logit[NEXP], nproj[NEXP];
#pragma unroll
  for (int e = 0; e < NEXP; ++e) { logit[e] = b_route[e]; nproj[e] = b_noise[e]; }
  const _Float16* x = seq + (size_t)t * DDIM;
  for (int k = 0; k < DDIM; ++k) {
    float xv = (float)x[k];
#pragma unroll
    for (int e = 0; e < NEXP; ++e) {
      logit[e] += xv * W_route[e * DDIM + k];
      nproj[e] += xv * W_noise[e * DDIM + k];
    }
  }
  float noisy[NEXP];
#pragma unroll
  for (int e = 0; e < NEXP; ++e)
    noisy[e] = logit[e] + noise[(size_t)t * NEXP + e] * softplusf_(nproj[e]);

  // top-2 of noisy
  int i0 = 0;
#pragma unroll
  for (int e = 1; e < NEXP; ++e) if (noisy[e] > noisy[i0]) i0 = e;
  int i1 = (i0 == 0) ? 1 : 0;
#pragma unroll
  for (int e = 0; e < NEXP; ++e)
    if (e != i0 && noisy[e] > noisy[i1]) i1 = e;
  float e1 = __expf(noisy[i1] - noisy[i0]);  // softmax over the two values
  float g0 = 1.0f / (1.0f + e1);
  float g1 = e1 / (1.0f + e1);

  const float mask = (l < nseg[b]) ? 1.0f : 0.0f;
  float g[NEXP];
#pragma unroll
  for (int e = 0; e < NEXP; ++e) g[e] = 0.0f;
  g[i0] = g0 * mask;  // mask folded: moe is only used through the masked sum
  g[i1] = g1 * mask;
#pragma unroll
  for (int e = 0; e < NEXP; ++e) gating[(size_t)t * NEXP + e] = g[e];

  // soft_gate = softmax(logits); masked partial sums for load balance
  float mx = logit[0];
#pragma unroll
  for (int e = 1; e < NEXP; ++e) mx = fmaxf(mx, logit[e]);
  float den = 0.0f;
  float ex[NEXP];
#pragma unroll
  for (int e = 0; e < NEXP; ++e) { ex[e] = __expf(logit[e] - mx); den += ex[e]; }
#pragma unroll
  for (int e = 0; e < NEXP; ++e)
    smbuf[threadIdx.x * NEXP + e] = mask * ex[e] / den;
  __syncthreads();
  if (threadIdx.x < NEXP) {
    float s = 0.0f;
    for (int i = 0; i < 256; ++i) s += smbuf[i * NEXP + threadIdx.x];
    loadpart[blockIdx.x * NEXP + threadIdx.x] = s;
  }
}

// ---------------------------------------------------------------------------
// MoE: one block per 16-token tile (tiles never straddle batch rows since
// L % 16 == 0). Computes all 8 experts densely with WMMA, weights by the
// (masked) gate, reduces over the 16 tokens in-wave and writes per-tile
// partial sums. Fully deterministic — no atomics.
// ---------------------------------------------------------------------------
__global__ __launch_bounds__(256) void moe_kernel(
    const _Float16* __restrict__ seq,  // [T,128] f16
    const _Float16* __restrict__ We1,  // [8,512,128] f16
    const _Float16* __restrict__ We2,  // [8,128,512] f16
    const float* __restrict__ be1, const float* __restrict__ be2,
    const float* __restrict__ gating,  // [T,8] (masked)
    float* __restrict__ part)          // [T/16, 128] tile partial sums
{
  __shared__ __align__(16) _Float16 a_lds[16 * DDIM];
  __shared__ __align__(16) _Float16 h_lds[16 * 512];
  __shared__ float gate_s[16 * NEXP];

  const int wave = threadIdx.x >> 5;
  const int lane = threadIdx.x & 31;
  const int l15  = lane & 15;
  const int half16 = (lane >> 4) << 3;
  const int t0 = blockIdx.x * 16;
  const int col = wave * 16 + l15;

  // Stage A tile (16 consecutive token rows are contiguous) + gates
  {
    const v8h* src = reinterpret_cast<const v8h*>(seq + (size_t)t0 * DDIM);
    v8h* dst = reinterpret_cast<v8h*>(a_lds);
    dst[threadIdx.x] = src[threadIdx.x];  // 16*128/8 == 256 vectors
    if (threadIdx.x < 16 * NEXP)
      gate_s[threadIdx.x] = gating[(size_t)t0 * NEXP + threadIdx.x];
  }
  __syncthreads();

  const _Float16* arow = &a_lds[l15 * DDIM + half16];
  v16h a0 = load_frag16(arow + 0);
  v16h a1 = load_frag16(arow + 32);
  v16h a2 = load_frag16(arow + 64);
  v16h a3 = load_frag16(arow + 96);

  v8f moe = {};
  for (int e = 0; e < NEXP; ++e) {
    const _Float16* W1 = We1 + (size_t)e * 512 * DDIM;
    const _Float16* W2 = We2 + (size_t)e * DDIM * 512;
    // GEMM1: hexp = relu(x @ We1.T + be1)   (16 x 512, K=128)
#pragma unroll
    for (int s = 0; s < 4; ++s) {
      const int n0 = wave * 64 + s * 16;
      const _Float16* pb = W1 + (size_t)(n0 + l15) * DDIM + half16;
      v8f acc = {};
      acc = wmma16(a0, load_frag16(pb + 0),  acc);
      acc = wmma16(a1, load_frag16(pb + 32), acc);
      acc = wmma16(a2, load_frag16(pb + 64), acc);
      acc = wmma16(a3, load_frag16(pb + 96), acc);
      const float bb = be1[e * 512 + n0 + l15];
#pragma unroll
      for (int j = 0; j < 8; ++j) {
        float v = acc[j] + bb;
        h_lds[(j + half16) * 512 + n0 + l15] = (_Float16)fmaxf(v, 0.0f);
      }
    }
    __syncthreads();
    // GEMM2: yexp = hexp @ We2.T + be2   (16 x 128, K=512)
    {
      const _Float16* hrow = &h_lds[l15 * 512 + half16];
      const _Float16* pb2  = W2 + (size_t)col * 512 + half16;
      v8f acc = {};
#pragma unroll
      for (int kk = 0; kk < 16; ++kk)
        acc = wmma16(load_frag16(hrow + kk * 32), load_frag16(pb2 + kk * 32), acc);
      const float b2 = be2[e * DDIM + col];
#pragma unroll
      for (int j = 0; j < 8; ++j) {
        const int m = j + half16;
        moe[j] += gate_s[m * NEXP + e] * (acc[j] + b2);
      }
    }
    __syncthreads();  // before next expert overwrites h_lds
  }

  // Reduce the 16 token rows of this tile (mask already in gates)
  float s = 0.0f;
#pragma unroll
  for (int j = 0; j < 8; ++j) s += moe[j];
  s += __shfl_xor(s, 16, 32);  // rows m and m+8 live in lane ^ 16 (wave32)
  if (lane < 16) part[(size_t)blockIdx.x * DDIM + col] = s;
}

__global__ __launch_bounds__(128) void sumpart_kernel(
    const float* __restrict__ part, float* __restrict__ seq_sum, int tilesPerB) {
  const int b = blockIdx.x, d = threadIdx.x;
  float s = 0.0f;
  for (int i = 0; i < tilesPerB; ++i)
    s += part[((size_t)b * tilesPerB + i) * DDIM + d];
  seq_sum[(size_t)b * DDIM + d] = s;
}

// ---------------------------------------------------------------------------
// Heads: fuse -> relu(bn) -> scalar out, per (r, b). Output order is
// (outs[0], outs[2], outs[1], lb).
// ---------------------------------------------------------------------------
__global__ __launch_bounds__(128) void head_kernel(
    const float* __restrict__ deep, const float* __restrict__ seq_sum,
    const float* __restrict__ Wd, const float* __restrict__ Wr,
    const float* __restrict__ Wo1, const float* __restrict__ bo1,
    const float* __restrict__ g_r, const float* __restrict__ beta_r,
    const float* __restrict__ Wo2, const float* __restrict__ bo2,
    float* __restrict__ out) {
  const int r = blockIdx.x / BDIM;
  const int b = blockIdx.x % BDIM;
  __shared__ float fuse_s[DDIM];
  __shared__ float red[DDIM];
  const int o = threadIdx.x;

  const float* wd = Wd + ((size_t)r * DDIM + o) * DDIM;
  const float* wr = Wr + ((size_t)r * DDIM + o) * DDIM;
  const float* dp = deep + (size_t)b * DDIM;
  const float* ss = seq_sum + (size_t)b * DDIM;
  float f = 0.0f;
  for (int k = 0; k < DDIM; ++k) f += dp[k] * wd[k] + ss[k] * wr[k];
  fuse_s[o] = f;
  __syncthreads();

  const float* w1 = Wo1 + ((size_t)r * DDIM + o) * DDIM;
  float h = bo1[r * DDIM + o];
  for (int k = 0; k < DDIM; ++k) h += fuse_s[k] * w1[k];
  const float inv = rsqrtf(1.0f + 1e-5f);
  h = h * (g_r[r * DDIM + o] * inv) + beta_r[r * DDIM + o];
  h = fmaxf(h, 0.0f);
  red[o] = h * Wo2[(size_t)r * DDIM + o];
  __syncthreads();
  for (int s = 64; s > 0; s >>= 1) {
    if (o < s) red[o] += red[o + s];
    __syncthreads();
  }
  if (o == 0) {
    const int slot = (r == 0) ? 0 : ((r == 2) ? 1 : 2);
    out[slot * BDIM + b] = red[0] + bo2[r];
  }
}

__global__ void lb_kernel(const float* __restrict__ loadpart, int nblk,
                          float* __restrict__ out) {
  if (threadIdx.x == 0 && blockIdx.x == 0) {
    float load[NEXP];
    for (int e = 0; e < NEXP; ++e) {
      float s = 0.0f;
      for (int i = 0; i < nblk; ++i) s += loadpart[i * NEXP + e];
      load[e] = s;
    }
    float tot = 0.0f;
    for (int e = 0; e < NEXP; ++e) tot += load[e];
    float lb = 0.0f;
    for (int e = 0; e < NEXP; ++e) {
      float n = load[e] / (tot + 1e-9f);
      lb += n * __logf(n * (float)NEXP + 1e-9f);
    }
    out[0] = lb;
  }
}

// ---------------------------------------------------------------------------
// Host launcher
// ---------------------------------------------------------------------------
extern "C" void kernel_launch(void* const* d_in, const int* in_sizes, int n_in,
                              void* d_out, int out_size, void* d_ws, size_t ws_size,
                              hipStream_t stream) {
  (void)in_sizes; (void)n_in; (void)out_size; (void)ws_size;
  const int Bn = BDIM, Ln = LDIM, T = TTOK;

  const int*   xs       = (const int*)d_in[0];
  const int*   nseg     = (const int*)d_in[2];
  const int*   ts       = (const int*)d_in[3];
  const int*   od       = (const int*)d_in[4];
  const float* noise    = (const float*)d_in[5];
  const float* seg_emb  = (const float*)d_in[6];
  const float* node_emb = (const float*)d_in[7];
  const float* slice_emb= (const float*)d_in[8];
  const float* W_all    = (const float*)d_in[9];
  const float* b_all    = (const float*)d_in[10];
  const float* Wih0     = (const float*)d_in[11];
  const float* Whh0     = (const float*)d_in[12];
  const float* bih0     = (const float*)d_in[13];
  const float* bhh0     = (const float*)d_in[14];
  const float* Wih1     = (const float*)d_in[15];
  const float* Whh1     = (const float*)d_in[16];
  const float* bih1     = (const float*)d_in[17];
  const float* bhh1     = (const float*)d_in[18];
  const float* W_dm     = (const float*)d_in[19];
  const float* b_dm     = (const float*)d_in[20];
  const float* g_dm     = (const float*)d_in[21];
  const float* beta_dm  = (const float*)d_in[22];
  const float* W_route  = (const float*)d_in[23];
  const float* b_route  = (const float*)d_in[24];
  const float* W_noise  = (const float*)d_in[25];
  const float* b_noise  = (const float*)d_in[26];
  const float* We1      = (const float*)d_in[27];
  const float* be1      = (const float*)d_in[28];
  const float* We2      = (const float*)d_in[29];
  const float* be2      = (const float*)d_in[30];
  const float* Wd       = (const float*)d_in[31];
  const float* Wr       = (const float*)d_in[32];
  const float* Wo1      = (const float*)d_in[33];
  const float* bo1      = (const float*)d_in[34];
  const float* g_r      = (const float*)d_in[35];
  const float* beta_r   = (const float*)d_in[36];
  const float* Wo2      = (const float*)d_in[37];
  const float* bo2      = (const float*)d_in[38];

  char* ws = (char*)d_ws;
  size_t off = 0;
  auto alloc = [&](size_t bytes) {
    size_t o = off;
    off = (off + bytes + 255) & ~(size_t)255;
    return o;
  };
  _Float16* rec_h  = (_Float16*)(ws + alloc((size_t)T * DDIM * 2));
  _Float16* h1_h   = (_Float16*)(ws + alloc((size_t)T * DDIM * 2));
  _Float16* seq_h  = (_Float16*)(ws + alloc((size_t)T * DDIM * 2));
  _Float16* Wih0h  = (_Float16*)(ws + alloc(512 * 128 * 2));
  _Float16* Whh0h  = (_Float16*)(ws + alloc(512 * 128 * 2));
  _Float16* Wih1h  = (_Float16*)(ws + alloc(512 * 128 * 2));
  _Float16* Whh1h  = (_Float16*)(ws + alloc(512 * 128 * 2));
  _Float16* We1h   = (_Float16*)(ws + alloc((size_t)NEXP * 512 * 128 * 2));
  _Float16* We2h   = (_Float16*)(ws + alloc((size_t)NEXP * 128 * 512 * 2));
  float* bias0     = (float*)(ws + alloc(512 * 4));
  float* bias1     = (float*)(ws + alloc(512 * 4));
  float* gating    = (float*)(ws + alloc((size_t)T * NEXP * 4));
  float* loadpart  = (float*)(ws + alloc((T / 256) * NEXP * 4));
  float* part      = (float*)(ws + alloc((size_t)(T / 16) * DDIM * 4));
  float* seq_sum   = (float*)(ws + alloc((size_t)Bn * DDIM * 4));
  float* deep      = (float*)(ws + alloc((size_t)Bn * DDIM * 4));
  float* xwbuf     = (float*)(ws + alloc((size_t)(Bn / 16) * Ln * 512 * 16 * 4));
  float* outp      = (float*)d_out;

  // Weight precision conversion (f32 -> f16) + bias fusion
  const int wsz = 512 * 128;
  f2h_kernel<<<(wsz + 255) / 256, 256, 0, stream>>>(Wih0, Wih0h, wsz);
  f2h_kernel<<<(wsz + 255) / 256, 256, 0, stream>>>(Whh0, Whh0h, wsz);
  f2h_kernel<<<(wsz + 255) / 256, 256, 0, stream>>>(Wih1, Wih1h, wsz);
  f2h_kernel<<<(wsz + 255) / 256, 256, 0, stream>>>(Whh1, Whh1h, wsz);
  const int esz = NEXP * 512 * 128;
  f2h_kernel<<<(esz + 255) / 256, 256, 0, stream>>>(We1, We1h, esz);
  f2h_kernel<<<(esz + 255) / 256, 256, 0, stream>>>(We2, We2h, esz);
  addbias_kernel<<<2, 256, 0, stream>>>(bih0, bhh0, bias0, 512);
  addbias_kernel<<<2, 256, 0, stream>>>(bih1, bhh1, bias1, 512);

  // Embedding / input projections
  recin_kernel<<<T, 128, 0, stream>>>(xs, ts, seg_emb, slice_emb, W_all, b_all, rec_h);
  deep_kernel<<<Bn, 128, 0, stream>>>(od, ts, node_emb, W_dm, b_dm, g_dm, beta_dm, deep);

  // LSTM layer 0: parallel x-projection pre-GEMM, then sequential recurrence
  xw_kernel<<<dim3(Ln, Bn / 16), 256, 0, stream>>>(rec_h, Wih0h, xwbuf);
  lstm_seq_kernel<<<Bn / 16, 256, 0, stream>>>(xwbuf, Whh0h, bias0, h1_h);
  // LSTM layer 1
  xw_kernel<<<dim3(Ln, Bn / 16), 256, 0, stream>>>(h1_h, Wih1h, xwbuf);
  lstm_seq_kernel<<<Bn / 16, 256, 0, stream>>>(xwbuf, Whh1h, bias1, seq_h);

  // Router + dense deterministic MoE
  router_kernel<<<T / 256, 256, 0, stream>>>(seq_h, W_route, b_route, W_noise,
                                             b_noise, noise, nseg, gating, loadpart);
  moe_kernel<<<T / 16, 256, 0, stream>>>(seq_h, We1h, We2h, be1, be2, gating, part);
  sumpart_kernel<<<Bn, 128, 0, stream>>>(part, seq_sum, Ln / 16);

  // Heads + load-balance scalar
  head_kernel<<<3 * Bn, 128, 0, stream>>>(deep, seq_sum, Wd, Wr, Wo1, bo1, g_r,
                                          beta_r, Wo2, bo2, outp);
  lb_kernel<<<1, 32, 0, stream>>>(loadpart, T / 256, outp + 3 * Bn);
}
